// LSTMEncoder_29867202577004
// MI455X (gfx1250) — compile-verified, matrix operand order checked by reading, not run on hardware
//
#include <hip/hip_runtime.h>

typedef __attribute__((ext_vector_type(16))) _Float16 v16h;
typedef __attribute__((ext_vector_type(8)))  float    v8f;

#define BATCH 256
#define CHAN  512
#define HDIM  8
#define WDIM  64
#define HID   512
#define G4    2048   // 4*HID
#define KTOT  1024   // HID (h part) + CHAN (x part)

// GEMM tile config: 64(M) x 128(N) block tile, K-slab 32, 8 wave32s,
// each wave owns a 32x32 output tile -> 4 wmma per K-slab.
#define BM 64
#define BN 128
#define BK 32
#define NSTAGE (KTOT / BK)   // 32

// ---------------------------------------------------------------------------
// CDNA5 async copy: global -> LDS, 16B per lane, tracked by ASYNCcnt.
// LDS operand is the low 32 bits of the generic pointer (ISA: LDS generic
// address truncates to the LDS byte offset).
// ---------------------------------------------------------------------------
__device__ __forceinline__ void async_ld_b128(void* lds_ptr, const void* gptr) {
    unsigned l = (unsigned)(uintptr_t)lds_ptr;
    asm volatile("global_load_async_to_lds_b128 %0, %1, off"
                 :: "v"(l), "v"(gptr)
                 : "memory");
}
__device__ __forceinline__ void wait_async0() {
    asm volatile("s_wait_asynccnt 0" ::: "memory");
}

// ---------------------------------------------------------------------------
// Mean-pool over H, transpose to x[w][b][c] (c contiguous), cast to f16.
// ---------------------------------------------------------------------------
__global__ __launch_bounds__(256)
void pool_kernel(const float* __restrict__ feat, _Float16* __restrict__ xh) {
    __shared__ float tile[64][65];          // [c_local][w], padded vs bank conflicts
    const int b  = blockIdx.x;              // 0..255
    const int c0 = blockIdx.y * 64;         // 0,64,...,448
    const int t  = threadIdx.x;
    const int w  = t & 63;
    const int cq = t >> 6;                  // 0..3
    #pragma unroll 4
    for (int ci = 0; ci < 16; ++ci) {
        const int cl = cq * 16 + ci;
        const int c  = c0 + cl;
        const float* p = feat + ((size_t)b * CHAN + c) * (HDIM * WDIM) + w;
        float s = 0.f;
        #pragma unroll
        for (int h = 0; h < HDIM; ++h) s += p[h * WDIM];
        tile[cl][w] = s * (1.0f / HDIM);
    }
    __syncthreads();
    const int cl = t & 63;
    const int wq = t >> 6;                  // 0..3
    #pragma unroll 4
    for (int wi = 0; wi < 16; ++wi) {
        const int ww = wq * 16 + wi;
        xh[((size_t)ww * BATCH + b) * CHAN + c0 + cl] = (_Float16)tile[cl][ww];
    }
}

// ---------------------------------------------------------------------------
// One-shot prep: weights f32->f16, bias sum, zero h(f16) and c(f32).
// ---------------------------------------------------------------------------
__global__ __launch_bounds__(256)
void prep_kernel(const float* __restrict__ Wih, const float* __restrict__ Whh,
                 const float* __restrict__ bih, const float* __restrict__ bhh,
                 _Float16* __restrict__ WihH, _Float16* __restrict__ WhhH,
                 float* __restrict__ bsum, _Float16* __restrict__ hH,
                 float* __restrict__ cS) {
    const size_t nW = (size_t)G4 * HID;     // 1048576
    const size_t i  = (size_t)blockIdx.x * 256 + threadIdx.x;
    if (i < nW) {
        WihH[i] = (_Float16)Wih[i];
    } else if (i < 2 * nW) {
        WhhH[i - nW] = (_Float16)Whh[i - nW];
    } else if (i < 2 * nW + G4) {
        const size_t j = i - 2 * nW;
        bsum[j] = bih[j] + bhh[j];
    } else if (i < 2 * nW + G4 + (size_t)BATCH * HID) {
        hH[i - (2 * nW + G4)] = (_Float16)0.f;
    } else if (i < 2 * nW + G4 + 2 * (size_t)BATCH * HID) {
        cS[i - (2 * nW + G4 + (size_t)BATCH * HID)] = 0.f;
    }
}

// ---------------------------------------------------------------------------
// Per-step gates GEMM: gates[b][g] = bsum[g] + [h | x_t](b,:) . [Whh | Wih](g,:)
// M=256, N=2048, K=1024. Double-buffered LDS fed by async global->LDS copies;
// uniform (scalar-branch) source select per K-stage; 4 wmma per wave per slab.
// ---------------------------------------------------------------------------
__global__ __launch_bounds__(256)
void lstm_gates_kernel(const _Float16* __restrict__ hH,    // [256][512]
                       const _Float16* __restrict__ xh,    // [64][256][512]
                       const _Float16* __restrict__ WhhH,  // [2048][512]
                       const _Float16* __restrict__ WihH,  // [2048][512]
                       const float*    __restrict__ bsum,  // [2048]
                       float*          __restrict__ gates, // [256][2048]
                       int t) {
    __shared__ alignas(64) _Float16 As[2][BM][BK];   // 2 x 4KB
    __shared__ alignas(64) _Float16 Bs[2][BN][BK];   // 2 x 8KB

    const int tid  = threadIdx.x;
    const int m0   = blockIdx.x * BM;
    const int n0   = blockIdx.y * BN;
    const int wv   = tid >> 5;
    const int lane = tid & 31;
    const int half = lane >> 4;             // K-half selector per WMMA layout
    const int lr   = lane & 15;             // row/col within fragment
    const int wm   = (wv & 1) * 32;         // wave M offset within tile
    const int wn   = (wv >> 1) * 32;        // wave N offset within tile

    const _Float16* xt = xh + (size_t)t * BATCH * CHAN;

    // async-staging thread mapping
    const int arow  = tid >> 2;             // 64 rows
    const int akoff = (tid & 3) * 8;        // 4 x 8 halves (16B)
    const int brow  = tid >> 1;             // 128 rows
    const int bkoff = (tid & 1) * 16;       // 2 x 16 halves (2 x 16B)

    auto issue_stage = [&](int s, int bsel) {
        const int k = s * BK;               // uniform -> scalar branch below
        const _Float16* gA;
        const _Float16* gB;
        if (k < HID) {
            gA = hH   + (size_t)(m0 + arow) * HID  + k + akoff;
            gB = WhhH + (size_t)(n0 + brow) * HID  + k + bkoff;
        } else {
            gA = xt   + (size_t)(m0 + arow) * CHAN + (k - HID) + akoff;
            gB = WihH + (size_t)(n0 + brow) * HID  + (k - HID) + bkoff;
        }
        async_ld_b128(&As[bsel][arow][akoff], gA);
        async_ld_b128(&Bs[bsel][brow][bkoff], gB);
        async_ld_b128(&Bs[bsel][brow][bkoff + 8], gB + 8);
    };

    // Init accumulators with fused bias (free epilogue add).
    v8f c00, c01, c10, c11;
    const float bn0 = bsum[n0 + wn + lr];
    const float bn1 = bsum[n0 + wn + 16 + lr];
    #pragma unroll
    for (int r = 0; r < 8; ++r) { c00[r] = bn0; c01[r] = bn1; c10[r] = bn0; c11[r] = bn1; }

    issue_stage(0, 0);

    int buf = 0;
    for (int s = 0; s < NSTAGE; ++s) {
        wait_async0();                      // this wave's async data is in LDS
        __syncthreads();                    // everyone's data is in LDS
        if (s + 1 < NSTAGE) issue_stage(s + 1, buf ^ 1);

        const v16h a0 = *(const v16h*)&As[buf][wm + lr][16 * half];
        const v16h a1 = *(const v16h*)&As[buf][wm + 16 + lr][16 * half];
        const v16h b0 = *(const v16h*)&Bs[buf][wn + lr][16 * half];
        const v16h b1 = *(const v16h*)&Bs[buf][wn + 16 + lr][16 * half];
        c00 = __builtin_amdgcn_wmma_f32_16x16x32_f16(false, a0, false, b0, (short)0, c00, false, false);
        c01 = __builtin_amdgcn_wmma_f32_16x16x32_f16(false, a0, false, b1, (short)0, c01, false, false);
        c10 = __builtin_amdgcn_wmma_f32_16x16x32_f16(false, a1, false, b0, (short)0, c10, false, false);
        c11 = __builtin_amdgcn_wmma_f32_16x16x32_f16(false, a1, false, b1, (short)0, c11, false, false);
        buf ^= 1;
    }

    // Epilogue: C/D layout — VGPR r, lane l -> M = r + 8*(l/16), N = l%16
    #pragma unroll
    for (int r = 0; r < 8; ++r) {
        const int ma = m0 + wm + r + 8 * half;
        const int mb = ma + 16;
        gates[(size_t)ma * G4 + n0 + wn + lr]      = c00[r];
        gates[(size_t)ma * G4 + n0 + wn + 16 + lr] = c01[r];
        gates[(size_t)mb * G4 + n0 + wn + lr]      = c10[r];
        gates[(size_t)mb * G4 + n0 + wn + 16 + lr] = c11[r];
    }
}

// ---------------------------------------------------------------------------
// Per-step cell update. Writes c (f32 ws), h (f16 ws for next GEMM), and
// streams h,c to d_out (final step's values survive).
// ---------------------------------------------------------------------------
__global__ __launch_bounds__(256)
void lstm_pointwise_kernel(const float* __restrict__ gates,
                           float* __restrict__ cS,
                           _Float16* __restrict__ hH,
                           float* __restrict__ outH,
                           float* __restrict__ outC) {
    const int idx = blockIdx.x * 256 + threadIdx.x;   // 0..131071
    const int b = idx >> 9;
    const int j = idx & 511;
    const float* g = gates + (size_t)b * G4;
    const float xi = g[j];
    const float xf = g[HID + j];
    const float xg = g[2 * HID + j];
    const float xo = g[3 * HID + j];
    const float si = 1.f / (1.f + __expf(-xi));
    const float sf = 1.f / (1.f + __expf(-xf));
    const float tg = tanhf(xg);
    const float so = 1.f / (1.f + __expf(-xo));
    const float cn = sf * cS[idx] + si * tg;
    const float hn = so * tanhf(cn);
    cS[idx]  = cn;
    hH[idx]  = (_Float16)hn;
    outH[idx] = hn;
    outC[idx] = cn;
}

// ---------------------------------------------------------------------------
// Workspace layout (all offsets 256B aligned):
//   x_f16   [64][256][512]  f16   @ 0          (16,777,216 B)
//   Wih_f16 [2048][512]     f16   @ 16777216   ( 2,097,152 B)
//   Whh_f16 [2048][512]     f16   @ 18874368   ( 2,097,152 B)
//   bsum    [2048]          f32   @ 20971520   (     8,192 B)
//   h_f16   [256][512]      f16   @ 20979712   (   262,144 B)
//   c       [256][512]      f32   @ 21241856   (   524,288 B)
//   gates   [256][2048]     f32   @ 21766144   ( 2,097,152 B)
// total ~= 23.9 MB
// ---------------------------------------------------------------------------
extern "C" void kernel_launch(void* const* d_in, const int* in_sizes, int n_in,
                              void* d_out, int out_size, void* d_ws, size_t ws_size,
                              hipStream_t stream) {
    const float* feat = (const float*)d_in[0];
    const float* Wih  = (const float*)d_in[1];
    const float* Whh  = (const float*)d_in[2];
    const float* bih  = (const float*)d_in[3];
    const float* bhh  = (const float*)d_in[4];

    char* ws = (char*)d_ws;
    _Float16* xh    = (_Float16*)(ws + 0);
    _Float16* WihH  = (_Float16*)(ws + 16777216);
    _Float16* WhhH  = (_Float16*)(ws + 18874368);
    float*    bsum  = (float*)   (ws + 20971520);
    _Float16* hH    = (_Float16*)(ws + 20979712);
    float*    cS    = (float*)   (ws + 21241856);
    float*    gates = (float*)   (ws + 21766144);

    float* outH = (float*)d_out;
    float* outC = outH + (size_t)BATCH * HID;

    pool_kernel<<<dim3(BATCH, CHAN / 64), 256, 0, stream>>>(feat, xh);

    // 2*1048576 + 2048 + 2*131072 = 2,361,344 elements -> 9224 blocks of 256
    prep_kernel<<<dim3(9224), 256, 0, stream>>>(Wih, Whh, bih, bhh,
                                                WihH, WhhH, bsum, hH, cS);

    for (int t = 0; t < WDIM; ++t) {
        lstm_gates_kernel<<<dim3(BATCH / BM, G4 / BN), 256, 0, stream>>>(
            hH, xh, WhhH, WihH, bsum, gates, t);
        lstm_pointwise_kernel<<<dim3((BATCH * HID) / 256), 256, 0, stream>>>(
            gates, cS, hH, outH, outC);
    }
}
